// SimpleCTRNN_61366492725329
// MI455X (gfx1250) — matrix operand order repaired and checked by reading, not compiled, not run on hardware
//
#include <hip/hip_runtime.h>

// ---------------- problem sizes ----------------
#define B_SZ  32
#define S_SZ  512
#define I_DIM 1024
#define H_DIM 1024

// ---------------- WMMA types -------------------
typedef __attribute__((ext_vector_type(16))) __bf16 v16bf;
typedef __attribute__((ext_vector_type(8)))  float  v8f;
// TDM descriptor groups (clang-23 6-arg builtin)
typedef __attribute__((ext_vector_type(4))) unsigned int v4u;
typedef __attribute__((ext_vector_type(8))) int          v8i;
typedef __attribute__((ext_vector_type(4))) int          v4i;

union Frag {
  v16bf v;
  uint4 q[2];
};

__device__ __forceinline__ unsigned short f2bf(float f) {
  union { float f; unsigned u; } c;
  c.f = f;
  unsigned r = c.u + 0x7FFFu + ((c.u >> 16) & 1u);   // round-to-nearest-even
  return (unsigned short)(r >> 16);
}

// =========================================================================
// Kernel 1: input projection  xw[b,s,:] = x[b,s,:] @ Wi + bias   (bf16 WMMA)
// Block tile 128(M) x 64(N), K-chunks of 32; 8 waves, each 32x32 (2x2 frags).
// =========================================================================
#define PBM 128
#define PBN 64
#define PBK 32
#define SA_STRIDE 48   // ushorts per row (32 + pad), 96B = 16B multiple
#define SB_STRIDE 48

__global__ __launch_bounds__(256) void ctrnn_proj_wmma(
    const float* __restrict__ x, const float* __restrict__ wi,
    const float* __restrict__ bias, float* __restrict__ out)
{
  __shared__ unsigned short sA[PBM * SA_STRIDE];   // [row][k]        (bf16 bits)
  __shared__ unsigned short sB[PBN * SB_STRIDE];   // [col][k] transposed

  const int t    = threadIdx.x;
  const int lane = t & 31;
  const int wave = t >> 5;
  const int l15  = lane & 15;
  const int lhi  = lane >> 4;

  const int mBase = (int)blockIdx.x * PBM;         // 128 blocks in M
  const int nBase = (int)blockIdx.y * PBN;         // 16 blocks in N
  const int wm = (wave & 3) * 32;                  // 4 waves in M
  const int wn = (wave >> 2) * 32;                 // 2 waves in N

  v8f acc[2][2] = {};

  for (int kc = 0; kc < I_DIM; kc += PBK) {
    // stage A: 128x32 f32 -> bf16 LDS (coalesced float4 loads)
#pragma unroll
    for (int i = 0; i < 4; ++i) {
      int idx = i * 256 + t;                       // 0..1023 float4s
      int row = idx >> 3;                          // 8 float4 per row
      int kq  = (idx & 7) * 4;
      float4 v = *(const float4*)(x + (size_t)(mBase + row) * I_DIM + (kc + kq));
      unsigned short* p = &sA[row * SA_STRIDE + kq];
      p[0] = f2bf(v.x); p[1] = f2bf(v.y); p[2] = f2bf(v.z); p[3] = f2bf(v.w);
    }
    // stage B transposed: Wi[kc+kr][nBase+nc] -> sB[nc][kr]
#pragma unroll
    for (int i = 0; i < 8; ++i) {
      int idx = i * 256 + t;                       // 0..2047
      int kr  = idx >> 6;
      int nc  = idx & 63;
      sB[nc * SB_STRIDE + kr] =
          f2bf(wi[(size_t)(kc + kr) * H_DIM + (nBase + nc)]);
    }
    __syncthreads();

    // fragments per ISA layout
    Frag fa[2], fb[2];
#pragma unroll
    for (int mt = 0; mt < 2; ++mt) {
      int row  = wm + mt * 16 + l15;
      int koff = lhi * 8;
      fa[mt].q[0] = *(const uint4*)&sA[row * SA_STRIDE + koff];
      fa[mt].q[1] = *(const uint4*)&sA[row * SA_STRIDE + koff + 16];
    }
#pragma unroll
    for (int nt = 0; nt < 2; ++nt) {
      int col   = wn + nt * 16 + l15;
      int kbase = lhi * 8;
      fb[nt].q[0] = *(const uint4*)&sB[col * SB_STRIDE + kbase];
      fb[nt].q[1] = *(const uint4*)&sB[col * SB_STRIDE + kbase + 16];
    }
#pragma unroll
    for (int mt = 0; mt < 2; ++mt)
#pragma unroll
      for (int nt = 0; nt < 2; ++nt)
        acc[mt][nt] = __builtin_amdgcn_wmma_f32_16x16x32_bf16(
            false, fa[mt].v, false, fb[nt].v, (short)0, acc[mt][nt],
            false, false);
    __syncthreads();
  }

  // epilogue: + bias, store f32
#pragma unroll
  for (int mt = 0; mt < 2; ++mt) {
#pragma unroll
    for (int nt = 0; nt < 2; ++nt) {
      int col  = nBase + wn + nt * 16 + l15;
      float bv = bias[col];
#pragma unroll
      for (int e = 0; e < 8; ++e) {
        int row = mBase + wm + mt * 16 + e + lhi * 8;
        out[(size_t)row * H_DIM + col] = acc[mt][nt][e] + bv;
      }
    }
  }
}

// =========================================================================
// Kernel 2: persistent recurrent scan. 16 blocks x 256 thr, block owns 64
// columns; W-slice (1024x64 bf16, col-major) lives in LDS for all 512 steps.
// h: f32 master + bf16 mirror in workspace. Per step, the Tensor Data Mover
// DMAs the 32x1024 bf16 h-mirror into LDS (iterate mode adds the 16B row pad
// so ds_load_b128 frag reads stay aligned & bank-spread). Grid barrier between
// steps via monotonic atomics in workspace.
// =========================================================================
#define RNB   16
#define RCOLS 64
#define SW_STRIDE 1032   // 1024 + 8 pad (2064B, 16B multiple)
#define SH_STRIDE 1032

__device__ __forceinline__ void grid_barrier(unsigned* bar, unsigned nblocks,
                                             unsigned phase) {
  __threadfence();
  __syncthreads();
  if (threadIdx.x == 0) {
    unsigned old = atomicAdd(&bar[0], 1u);         // monotonic arrival count
    if (old == nblocks * (phase + 1u) - 1u) {
      atomicAdd(&bar[1], 1u);                      // gen -> phase+1
    } else {
      while (*(volatile unsigned*)&bar[1] <= phase) {
        __builtin_amdgcn_s_sleep(2);
      }
    }
  }
  __syncthreads();
  __threadfence();
}

extern __shared__ __align__(16) unsigned short rnn_smem[];

__global__ __launch_bounds__(256) void ctrnn_rnn_wmma(
    const float* __restrict__ w, const float* __restrict__ tau,
    float* __restrict__ out, float* __restrict__ g_h,
    unsigned short* __restrict__ g_hbf, unsigned* __restrict__ bar)
{
  unsigned short* sW = rnn_smem;                        // [64][1032]
  unsigned short* sH = rnn_smem + RCOLS * SW_STRIDE;    // [32][1032]

  const int t    = threadIdx.x;
  const int lane = t & 31;
  const int wave = t >> 5;
  const int l15  = lane & 15;
  const int lhi  = lane >> 4;
  const int nBase = (int)blockIdx.x * RCOLS;
  const int mt = wave & 1;        // batch tile: rows 0-15 / 16-31
  const int nt = wave >> 1;       // column tile 0..3

  // one-time: preload W slice, transposed+converted: W[k][nBase+c] -> sW[c][k]
#pragma unroll 4
  for (int i = 0; i < 256; ++i) {                  // 65536 elems / 256 thr
    int idx = i * 256 + t;
    int k = idx >> 6;
    int c = idx & 63;
    sW[c * SW_STRIDE + k] = f2bf(w[(size_t)k * H_DIM + nBase + c]);
  }
  // zero this block's h slice (h0 = 0)
#pragma unroll
  for (int i = 0; i < 8; ++i) {
    int idx = i * 256 + t;                         // 0..2047
    int b = idx >> 6, c = idx & 63;
    g_h[(size_t)b * H_DIM + nBase + c]   = 0.0f;
    g_hbf[(size_t)b * H_DIM + nBase + c] = 0;
  }

  // ---- build TDM descriptor (D#) for h-mirror -> sH, per ISA ch.8 ----
  // 2D: tile 1024x1 bf16, iterate 32x; global += 1024 elems, LDS += 1032 elems
  const unsigned lds_off = (unsigned)(unsigned long long)(void*)sH;
  const unsigned long long ga = (unsigned long long)g_hbf;
  v4u g0;
  g0[0] = 1u;                                        // count=1, user mode
  g0[1] = lds_off;                                   // lds_addr
  g0[2] = (unsigned)ga;                              // global_addr[31:0]
  g0[3] = ((unsigned)(ga >> 32) & 0x01FFFFFFu)       // global_addr[56:32]
          | (2u << 30);                              // type=2 ("image")
  v8i g1;
  g1[0] = (int)((1u << 16) | (1u << 19));            // data_size=2B, iterate_en
  g1[1] = (int)((1024u & 0xFFFFu) << 16);            // tensor_dim0[15:0]
  g1[2] = (int)((1024u >> 16) | ((32u & 0xFFFFu) << 16)); // td0 hi | td1 lo
  g1[3] = (int)((32u >> 16) | (1024u << 16));        // td1 hi | tile_dim0=1024
  g1[4] = (int)1u;                                   // tile_dim1=1, tile_dim2=0
  g1[5] = (int)1024;                                 // tensor_dim0_stride lo32
  g1[6] = 0;                                         // stride hi | td1_stride lo
  g1[7] = 0;                                         // td1_stride hi (unused)
  v4i g2;
  g2[0] = 1;                                         // tensor_dim2
  g2[1] = 1032;                                      // lds_addr_increment
  g2[2] = 1024;                                      // global_addr_increment
  g2[3] = (int)(31u << 16);                          // iterate_count=31 (32x)
  v4i g3 = {0, 0, 0, 0};
  v8i g4 = {0, 0, 0, 0, 0, 0, 0, 0};                 // extra group (unused)

  grid_barrier(bar, RNB, 0u);

  const int   col  = nBase + nt * 16 + l15;
  const float itau = 1.0f / tau[col];

  for (int step = 0; step < S_SZ; ++step) {
    // stage full h (bf16, 64KB) into LDS via Tensor Data Mover (wave 0 only:
    // EXEC is ignored by TDM, one DMA per instruction)
    if (t < 32) {
      __builtin_amdgcn_tensor_load_to_lds(g0, g1, g2, g3, g4, 0);
      __builtin_amdgcn_s_wait_tensorcnt(0);
    }
    __syncthreads();

    v8f acc = {};
#pragma unroll 4
    for (int kc = 0; kc < H_DIM; kc += 32) {
      Frag fa, fb;
      {
        int row  = mt * 16 + l15;
        int koff = kc + lhi * 8;
        fa.q[0] = *(const uint4*)&sH[row * SH_STRIDE + koff];
        fa.q[1] = *(const uint4*)&sH[row * SH_STRIDE + koff + 16];
      }
      {
        int c     = nt * 16 + l15;
        int kbase = kc + lhi * 8;
        fb.q[0] = *(const uint4*)&sW[c * SW_STRIDE + kbase];
        fb.q[1] = *(const uint4*)&sW[c * SW_STRIDE + kbase + 16];
      }
      acc = __builtin_amdgcn_wmma_f32_16x16x32_bf16(
          false, fa.v, false, fb.v, (short)0, acc, false, false);
    }

    // epilogue: h_new = h + itau*(relu(acc + xt) - h); ys[b,t] = h_new
#pragma unroll
    for (int e = 0; e < 8; ++e) {
      int b = mt * 16 + e + lhi * 8;
      size_t oidx = ((size_t)b * S_SZ + step) * H_DIM + col;
      float xt   = out[oidx];                      // xw (in-place buffer)
      float pre  = acc[e] + xt;
      float r    = pre > 0.0f ? pre : 0.0f;
      size_t hix = (size_t)b * H_DIM + col;
      float hold = g_h[hix];
      float hnew = hold + itau * (r - hold);
      g_h[hix]   = hnew;
      g_hbf[hix] = f2bf(hnew);
      out[oidx]  = hnew;
      if (step + 1 < S_SZ) {
        __builtin_prefetch(&out[oidx + H_DIM], 0, 1);  // next step's xt
      }
    }
    grid_barrier(bar, RNB, 1u + (unsigned)step);
  }
}

// =========================================================================
extern "C" void kernel_launch(void* const* d_in, const int* in_sizes, int n_in,
                              void* d_out, int out_size, void* d_ws,
                              size_t ws_size, hipStream_t stream) {
  const float* x    = (const float*)d_in[0];   // [B,S,I]
  const float* w    = (const float*)d_in[1];   // [H,H]
  const float* wi   = (const float*)d_in[2];   // [I,H]
  const float* bias = (const float*)d_in[3];   // [1,H]
  const float* tau  = (const float*)d_in[4];   // [1,H]
  float* out = (float*)d_out;                  // [B,S,H]

  // workspace: [0,256) barrier counters; then h f32 (128KB); then h bf16 (64KB)
  unsigned* bar = (unsigned*)d_ws;
  float* g_h = (float*)((char*)d_ws + 256);
  unsigned short* g_hbf =
      (unsigned short*)((char*)d_ws + 256 + (size_t)B_SZ * H_DIM * 4);

  (void)hipMemsetAsync(d_ws, 0, 256, stream);  // graph-capturable

  ctrnn_proj_wmma<<<dim3((B_SZ * S_SZ) / PBM, H_DIM / PBN), 256, 0, stream>>>(
      x, wi, bias, out);

  size_t smem = (size_t)(RCOLS * SW_STRIDE + B_SZ * SH_STRIDE) *
                sizeof(unsigned short);        // ~198KB < 320KB per WGP
  ctrnn_rnn_wmma<<<RNB, 256, smem, stream>>>(w, tau, out, g_h, g_hbf, bar);
}
